// CARAFE3D_77163382440813
// MI455X (gfx1250) — compile-verified, matrix operand order checked by reading, not run on hardware
//
#include <hip/hip_runtime.h>
#include <hip/hip_bf16.h>
#include <math.h>

typedef __attribute__((ext_vector_type(16))) __bf16 v16bf;
typedef __attribute__((ext_vector_type(8)))  __bf16 v8bf;
typedef __attribute__((ext_vector_type(8)))  float  v8f;
typedef unsigned int u32x4 __attribute__((ext_vector_type(4)));
typedef int          i32x8 __attribute__((ext_vector_type(8)));
typedef int          i32x4 __attribute__((ext_vector_type(4)));

#define C1 32
#define C2 16
#define CM 8          // C1/4
#define K3 27
#define R3 8
#define CK 216        // k3*r3 == cm*27
#define NB 2
#define HH 32
#define WW 32
#define DD 32
#define SP (HH*WW*DD)   // 32768
#define VOX (NB*SP)     // 65536
#define NTILES 14       // ceil(216/16)
#define KCH 7           // ceil(216/32)

// 16-bit A/B matrix VGPR layout (CDNA5 ISA §7.12.2):
// lane half 0 (lanes 0-15):  VGPR0-3 -> K 0..7,  VGPR4-7 -> K 16..23
// lane half 1 (lanes 16-31): VGPR0-3 -> K 8..15, VGPR4-7 -> K 24..31
__device__ __forceinline__ int wmma_k16(int i, int half) {
  int j = i >> 1, slot = i & 1;
  int base = (j < 4) ? 0 : 16;
  return base + half * 8 + (j & 3) * 2 + slot;
}

static __device__ __forceinline__ v8f wmma_bf16(v16bf a, v16bf b, v8f c) {
  return __builtin_amdgcn_wmma_f32_16x16x32_bf16(false, a, false, b, (short)0, c,
                                                 false, false);
}

// 1-D TDM copy (LDS <-> global), bf16 elements. D# per CDNA5 ISA:
// g0: count=1 | lds_addr | global_addr[56:0] | type=2 ("image")
// g1: data_size=1(2B), tensor_dim0=len, tensor_dim1=1, tile_dim0=len,
//     tile_dim1=0 (1-D), tensor_dim0_stride=len, workgroup_mask=0 (no cluster)
// 6-arg builtin form (clang-23 / therock-10.0): (g0, g1, g2, g3, g4, cpol)
__device__ __forceinline__ void tdm_copy_1d(bool store, unsigned lds_addr,
                                            unsigned long long gaddr,
                                            unsigned len) {
  u32x4 g0;
  g0[0] = 1u;                                            // count=1, user mode
  g0[1] = lds_addr;                                      // LDS byte address
  g0[2] = (unsigned)(gaddr & 0xFFFFFFFFu);               // global_addr[31:0]
  g0[3] = (unsigned)((gaddr >> 32) & 0x01FFFFFFu) | (2u << 30);  // [56:32]|type=2
  i32x8 g1;
  g1[0] = (int)(1u << 16);                               // data_size = 2 bytes
  g1[1] = (int)((len & 0xFFFFu) << 16);                  // tensor_dim0[15:0]
  g1[2] = (int)(((len >> 16) & 0xFFFFu) | (1u << 16));   // dim0[31:16]|tensor_dim1=1
  g1[3] = (int)((len & 0xFFFFu) << 16);                  // tile_dim0 = len
  g1[4] = 0;                                             // tile_dim1/2 = 0 (1-D)
  g1[5] = (int)len;                                      // tensor_dim0_stride lo
  g1[6] = 0;
  g1[7] = 0;
  i32x4 z4 = {};
  i32x8 z8 = {};
  if (store)
    __builtin_amdgcn_tensor_store_from_lds(g0, g1, z4, z4, z8, 0);
  else
    __builtin_amdgcn_tensor_load_to_lds(g0, g1, z4, z4, z8, 0);
}

// ---------------- Stage 0: pack w_enc (f32 [216][216]) into per-lane bf16 B-tiles
__global__ void __launch_bounds__(256)
k_pack_wenc(const float* __restrict__ w_enc, __bf16* __restrict__ wpack) {
  int tid = blockIdx.x * blockDim.x + threadIdx.x;   // one thread = one (nt,q,lane)
  if (tid >= NTILES * KCH * 32) return;
  int lane = tid & 31;
  int q    = (tid >> 5) % KCH;
  int nt   = tid / (32 * KCH);
  int half = lane >> 4, lidx = lane & 15;
  int oc = nt * 16 + lidx;
  __bf16* dst = wpack + ((size_t)tid << 4);
#pragma unroll
  for (int i = 0; i < 16; ++i) {
    int kk = q * 32 + wmma_k16(i, half);
    dst[i] = (oc < CK && kk < CK) ? (__bf16)w_enc[(size_t)oc * CK + kk]
                                  : (__bf16)0.0f;
  }
}

// ---------------- Stage 1: 1x1 down-conv  x[2,32,spatial] -> t[2,8,spatial] bf16
__global__ void __launch_bounds__(256)
k_down(const float* __restrict__ x, const float* __restrict__ w_down,
       const float* __restrict__ b_down, __bf16* __restrict__ t) {
  const int lane = threadIdx.x & 31;
  const int wv   = threadIdx.x >> 5;
  const int half = lane >> 4;
  const int lidx = lane & 15;
  const int voxBase = (blockIdx.x * (blockDim.x >> 5) + wv) * 16;

  v16bf a;
  {
    int vox = voxBase + lidx;
    int n = vox >> 15, s = vox & (SP - 1);
    const float* xp = x + (size_t)n * C1 * SP + s;
#pragma unroll
    for (int i = 0; i < 16; ++i) {
      int k = wmma_k16(i, half);
      a[i] = (__bf16)xp[(size_t)k * SP];
    }
  }
  v16bf b;
#pragma unroll
  for (int i = 0; i < 16; ++i) {
    int k = wmma_k16(i, half);
    b[i] = (lidx < CM) ? (__bf16)w_down[lidx * C1 + k] : (__bf16)0.0f;
  }
  v8f c = {};
  c = wmma_bf16(a, b, c);
  if (lidx < CM) {
    float bias = b_down[lidx];
#pragma unroll
    for (int j = 0; j < 8; ++j) {
      int m   = j + half * 8;
      int vox = voxBase + m;
      int n = vox >> 15, s = vox & (SP - 1);
      t[(size_t)n * CM * SP + (size_t)lidx * SP + s] = (__bf16)(c[j] + bias);
    }
  }
}

// ---------------- Stage 2: 3^3 conv 8->216 (+bias), softmax over k3 -> kern bf16
// 2 waves/block; result block streamed out via TDM (tensor_store_from_lds).
__global__ void __launch_bounds__(64)
k_enc(const __bf16* __restrict__ t, const __bf16* __restrict__ wpack,
      const float* __restrict__ b_enc, __bf16* __restrict__ kern) {
  __shared__ float  encb[2][16][224];     // 28.0 KB
  __shared__ __bf16 kbuf[2][16 * CK];     // 13.5 KB staging for TDM store
  const int lane = threadIdx.x & 31;
  const int wv   = threadIdx.x >> 5;
  const int half = lane >> 4;
  const int lidx = lane & 15;
  const int voxBase = (blockIdx.x * 2 + wv) * 16;

  __builtin_prefetch(wpack + ((size_t)lane << 4), 0, 1);   // global_prefetch_b8

  // Cache A for all 7 K-chunks (K = 216 = cm(8) outer x tap(27) inner)
  v16bf aC[KCH];
  {
    int vox = voxBase + lidx;
    int n = vox >> 15, rem = vox & (SP - 1);
    int h = rem >> 10, w = (rem >> 5) & 31, d = rem & 31;
    const __bf16* tb = t + (size_t)n * CM * SP;
#pragma unroll
    for (int q = 0; q < KCH; ++q) {
#pragma unroll
      for (int i = 0; i < 16; ++i) {
        int kk = q * 32 + wmma_k16(i, half);
        __bf16 v = (__bf16)0.0f;
        if (kk < CK) {
          int cm = kk / K3, tap = kk % K3;
          int kh = tap / 9, kw2 = (tap / 3) % 3, kd = tap % 3;
          int nh = h + kh - 1, nw = w + kw2 - 1, nd = d + kd - 1;
          if (nh >= 0 && nh < HH && nw >= 0 && nw < WW && nd >= 0 && nd < DD)
            v = tb[(size_t)cm * SP + nh * (WW * DD) + nw * DD + nd];
        }
        aC[q][i] = v;
      }
    }
  }

  // 14 N-tiles x 7 K-steps; B = packed bf16 tile (single 32B vector load)
  for (int nt = 0; nt < NTILES; ++nt) {
    v8f c = {};
#pragma unroll
    for (int q = 0; q < KCH; ++q) {
      v16bf b = *reinterpret_cast<const v16bf*>(
          wpack + (((size_t)(nt * KCH + q) * 32 + lane) << 4));
      c = wmma_bf16(aC[q], b, c);
    }
    int oc = nt * 16 + lidx;
    if (oc < CK) {
      float bias = b_enc[oc];
#pragma unroll
      for (int j = 0; j < 8; ++j) {
        int m = j + half * 8;
        encb[wv][m][oc] = c[j] + bias;
      }
    }
  }
  asm volatile("s_wait_dscnt 0" ::: "memory");  // wave-local LDS handoff

  // softmax over k3 per (voxel, r): normalize in place (disjoint slots per task)
  for (int tsk = lane; tsk < 128; tsk += 32) {
    int m = tsk >> 3, r = tsk & 7;
    float mx = -1e30f;
    for (int k = 0; k < K3; ++k) mx = fmaxf(mx, encb[wv][m][k * R3 + r]);
    float sum = 0.f;
    for (int k = 0; k < K3; ++k) sum += __expf(encb[wv][m][k * R3 + r] - mx);
    float inv = 1.f / sum;
    for (int k = 0; k < K3; ++k)
      encb[wv][m][k * R3 + r] = __expf(encb[wv][m][k * R3 + r] - mx) * inv;
  }
  asm volatile("s_wait_dscnt 0" ::: "memory");

  // pack bf16 into LDS staging (16B chunks), then one TDM store of 6912B
  for (int e = lane; e < 16 * K3; e += 32) {
    int m = e / K3, c8 = e % K3;
    v8bf v;
#pragma unroll
    for (int u = 0; u < 8; ++u) v[u] = (__bf16)encb[wv][m][c8 * 8 + u];
    *reinterpret_cast<v8bf*>(&kbuf[wv][m * CK + c8 * 8]) = v;
  }
  asm volatile("s_wait_dscnt 0" ::: "memory");
  tdm_copy_1d(true, (unsigned)(uintptr_t)&kbuf[wv][0],
              (unsigned long long)(uintptr_t)(kern + (size_t)voxBase * CK),
              16 * CK);
  __builtin_amdgcn_s_wait_tensorcnt(0);
}

// ---------------- Stage 3: reassembly + pixel-shuffle + 1x1 out-conv (fused)
// Per-wave kern row (432B) fetched by TDM into LDS, overlapped with w_out loads.
__global__ void __launch_bounds__(256)
k_carafe(const float* __restrict__ x, const __bf16* __restrict__ kern,
         const float* __restrict__ w_out, const float* __restrict__ b_out,
         float* __restrict__ out) {
  __shared__ __bf16 vbuf[8][C1][R3];   // per-wave V = P @ kern
  __shared__ __bf16 kbuf[8][CK];       // per-wave kern row via TDM
  const int lane = threadIdx.x & 31;
  const int wv   = threadIdx.x >> 5;
  const int half = lane >> 4;
  const int lidx = lane & 15;
  const int vox  = blockIdx.x * 8 + wv;
  const int n = vox >> 15, rem = vox & (SP - 1);
  const int h = rem >> 10, w = (rem >> 5) & 31, d = rem & 31;

  // kick off DMA of this voxel's kern row (overlaps the global loads below)
  tdm_copy_1d(false, (unsigned)(uintptr_t)&kbuf[wv][0],
              (unsigned long long)(uintptr_t)(kern + (size_t)vox * CK), CK);

  // constant A: w_out [16 x 32]
  v16bf aW;
#pragma unroll
  for (int i = 0; i < 16; ++i) {
    int k = wmma_k16(i, half);
    aW[i] = (__bf16)w_out[lidx * C1 + k];
  }
  __builtin_amdgcn_s_wait_tensorcnt(0);
  asm volatile("" ::: "memory");

  // B: kern tile [27(->32) x 8(->16)] from LDS
  v16bf bK;
#pragma unroll
  for (int i = 0; i < 16; ++i) {
    int k = wmma_k16(i, half);
    bK[i] = (lidx < R3 && k < K3) ? kbuf[wv][k * R3 + lidx] : (__bf16)0.0f;
  }
  // V = patches(32x27) @ kern(27x8): two M-tiles
  const float* xb = x + (size_t)n * C1 * SP;
#pragma unroll
  for (int t2 = 0; t2 < 2; ++t2) {
    int cch = lidx + t2 * 16;        // A row = input channel
    v16bf aP;
#pragma unroll
    for (int i = 0; i < 16; ++i) {
      int k = wmma_k16(i, half);
      __bf16 v = (__bf16)0.0f;
      if (k < K3) {
        int kh = k / 9, kw2 = (k / 3) % 3, kd = k % 3;
        int nh = h + kh - 1, nw = w + kw2 - 1, nd = d + kd - 1;
        if (nh >= 0 && nh < HH && nw >= 0 && nw < WW && nd >= 0 && nd < DD)
          v = (__bf16)xb[(size_t)cch * SP + nh * (WW * DD) + nw * DD + nd];
      }
      aP[i] = v;
    }
    v8f c = {};
    c = wmma_bf16(aP, bK, c);
    if (lidx < R3) {
#pragma unroll
      for (int j = 0; j < 8; ++j) {
        int m = j + half * 8 + t2 * 16;   // channel c
        vbuf[wv][m][lidx] = (__bf16)c[j];
      }
    }
  }
  asm volatile("s_wait_dscnt 0" ::: "memory");  // wave-local LDS transpose

  // Y = w_out(16x32) @ V(32x8)
  v16bf b2;
#pragma unroll
  for (int i = 0; i < 16; ++i) {
    int k = wmma_k16(i, half);
    b2[i] = (lidx < R3) ? vbuf[wv][k][lidx] : (__bf16)0.0f;
  }
  v8f y = {};
  y = wmma_bf16(aW, b2, y);
  if (lidx < R3) {
    int rh = lidx >> 2, rw = (lidx >> 1) & 1, rd = lidx & 1;
    int oh = h * 2 + rh, ow = w * 2 + rw, od = d * 2 + rd;
#pragma unroll
    for (int j = 0; j < 8; ++j) {
      int m = j + half * 8;   // c2
      size_t idx = ((((size_t)n * C2 + m) * (HH * 2) + oh) * (WW * 2) + ow) *
                       (DD * 2) + od;
      out[idx] = y[j] + b_out[m];
    }
  }
}

extern "C" void kernel_launch(void* const* d_in, const int* in_sizes, int n_in,
                              void* d_out, int out_size, void* d_ws, size_t ws_size,
                              hipStream_t stream) {
  const float* x      = (const float*)d_in[0];
  const float* w_down = (const float*)d_in[1];
  const float* b_down = (const float*)d_in[2];
  const float* w_enc  = (const float*)d_in[3];
  const float* b_enc  = (const float*)d_in[4];
  const float* w_out  = (const float*)d_in[5];
  const float* b_out  = (const float*)d_in[6];
  float* out = (float*)d_out;

  __bf16* t     = (__bf16*)d_ws;                                       // 1 MB
  __bf16* kern  = (__bf16*)((char*)d_ws + (size_t)VOX * CM * 2);       // 28.3 MB
  __bf16* wpack = (__bf16*)((char*)d_ws + (size_t)VOX * CM * 2
                                        + (size_t)VOX * CK * 2);       // 100 KB

  k_pack_wenc<<<(NTILES * KCH * 32 + 255) / 256, 256, 0, stream>>>(w_enc, wpack);
  k_down  <<<VOX / (16 * 8), 256, 0, stream>>>(x, w_down, b_down, t);
  k_enc   <<<VOX / (16 * 2), 64,  0, stream>>>(t, wpack, b_enc, kern);
  k_carafe<<<VOX / 8,        256, 0, stream>>>(x, kern, w_out, b_out, out);
}
